// MLP_learner_12309376271104
// MI455X (gfx1250) — matrix operand order repaired and checked by reading, not compiled, not run on hardware
//
#include <hip/hip_runtime.h>
#include <hip/hip_bf16.h>

#define NROWS 10000
#define ISIZE 256
#define KSEL  31
#define NPAD  10112      // 79 * 128
#define NTILE 79
#define TILE  128
#define LDK   40         // LDS row stride in halves (80B): 16B-aligned frags, bank-spread
#define KTILES (ISIZE / 32)

typedef _Float16 v8h  __attribute__((ext_vector_type(8)));
typedef _Float16 v16h __attribute__((ext_vector_type(16)));
typedef float    v8f  __attribute__((ext_vector_type(8)));
typedef int      v4i  __attribute__((__vector_size__(16)));

#if defined(__has_builtin)
#if __has_builtin(__builtin_amdgcn_global_load_async_to_lds_b128)
#define HAVE_ASYNC_LDS 1
#endif
#endif

static __device__ __forceinline__ void wait_async_le8() {
#if __has_builtin(__builtin_amdgcn_s_wait_asynccnt)
  __builtin_amdgcn_s_wait_asynccnt(8);
#else
  asm volatile("s_wait_asynccnt 0x8" ::: "memory");
#endif
}
static __device__ __forceinline__ void wait_async_le0() {
#if __has_builtin(__builtin_amdgcn_s_wait_asynccnt)
  __builtin_amdgcn_s_wait_asynccnt(0);
#else
  asm volatile("s_wait_asynccnt 0x0" ::: "memory");
#endif
}

// ---------------------------------------------------------------------------
// Kernel 1: h = relu(x@W1^T + b1) @ W2^T + b2; e = h / max(||h||, eps)
// emit e_hi = (f16)e, e_lo = (f16)(e - e_hi). Block = 256 thr, 16 rows/block.
// ---------------------------------------------------------------------------
__global__ __launch_bounds__(256) void mlp_norm_kernel(
    const float* __restrict__ feat, const float* __restrict__ W1,
    const float* __restrict__ b1,   const float* __restrict__ W2,
    const float* __restrict__ b2,
    _Float16* __restrict__ ehi, _Float16* __restrict__ elo) {
  __shared__ float xs[16][ISIZE];
  __shared__ float hs[16][ISIZE];
  __shared__ float red[16][16];
  __shared__ float nrm[16];
  const int t = threadIdx.x;
  const int bid = blockIdx.x;

  if (bid == NROWS / 16) {  // zero the padding rows of e_hi / e_lo
    const int padElems = (NPAD - NROWS) * ISIZE;
    for (int i = t; i < padElems; i += 256) {
      ehi[(size_t)NROWS * ISIZE + i] = (_Float16)0.f;
      elo[(size_t)NROWS * ISIZE + i] = (_Float16)0.f;
    }
    return;
  }

  const int row0 = bid * 16;
#pragma unroll
  for (int r = 0; r < 16; ++r) xs[r][t] = feat[(size_t)(row0 + r) * ISIZE + t];
  __syncthreads();

  float acc[16];
  // layer 1: thread t computes output feature t for all 16 rows
  {
    const float bb = b1[t];
#pragma unroll
    for (int r = 0; r < 16; ++r) acc[r] = bb;
#pragma unroll 1
    for (int k = 0; k < ISIZE; ++k) {
      const float w = W1[(size_t)t * ISIZE + k];
#pragma unroll
      for (int r = 0; r < 16; ++r) acc[r] = fmaf(w, xs[r][k], acc[r]);
    }
#pragma unroll
    for (int r = 0; r < 16; ++r) hs[r][t] = fmaxf(acc[r], 0.f);
  }
  __syncthreads();
  // layer 2
  {
    const float bb = b2[t];
#pragma unroll
    for (int r = 0; r < 16; ++r) acc[r] = bb;
#pragma unroll 1
    for (int k = 0; k < ISIZE; ++k) {
      const float w = W2[(size_t)t * ISIZE + k];
#pragma unroll
      for (int r = 0; r < 16; ++r) acc[r] = fmaf(w, hs[r][k], acc[r]);
    }
  }
  __syncthreads();
#pragma unroll
  for (int r = 0; r < 16; ++r) xs[r][t] = acc[r];
  __syncthreads();

  // row norms: 16 threads per row
  {
    const int rr = t >> 4, c = t & 15;
    float s = 0.f;
#pragma unroll
    for (int i = 0; i < 16; ++i) {
      const float v = xs[rr][c + 16 * i];
      s = fmaf(v, v, s);
    }
    red[rr][c] = s;
  }
  __syncthreads();
  if ((t & 15) == 0) {
    const int rr = t >> 4;
    float tot = 0.f;
#pragma unroll
    for (int i = 0; i < 16; ++i) tot += red[rr][i];
    nrm[rr] = fmaxf(sqrtf(tot), 1e-12f);
  }
  __syncthreads();

#pragma unroll
  for (int r = 0; r < 16; ++r) {
    const float e = xs[r][t] / nrm[r];
    const _Float16 hi = (_Float16)e;
    const _Float16 lo = (_Float16)(e - (float)hi);
    ehi[(size_t)(row0 + r) * ISIZE + t] = hi;
    elo[(size_t)(row0 + r) * ISIZE + t] = lo;
  }
}

// ---------------------------------------------------------------------------
// Fragment gathers per CDNA5 ISA VGPR layouts (wave32).
// A 16x32 f16: lanes<16 K = {0..7, 16..23}; lanes>=16 K = {8..15, 24..31}
// B 32x16 f16: lanes<16 K = {0..7, 16..23}; lanes>=16 K = {8..15, 24..31}
// ---------------------------------------------------------------------------
static __device__ __forceinline__ v16h frag_a(const _Float16* p) {
  const v8h lo = *(const v8h*)p;
  const v8h hi = *(const v8h*)(p + 16);
  return __builtin_shufflevector(lo, hi, 0, 1, 2, 3, 4, 5, 6, 7,
                                 8, 9, 10, 11, 12, 13, 14, 15);
}
static __device__ __forceinline__ v16h frag_b(const _Float16* p) {
  const v8h lo = *(const v8h*)p;
  const v8h hi = *(const v8h*)(p + 16);
  return __builtin_shufflevector(lo, hi, 0, 1, 2, 3, 4, 5, 6, 7,
                                 8, 9, 10, 11, 12, 13, 14, 15);
}

// ---------------------------------------------------------------------------
// Kernel 2: S = (ehi+elo) @ (ehi+elo)^T  via hi*hi + hi*lo + lo*hi (f32 acc).
// Block = 256 thr = 8 waves; 128x128 tile; each wave 32x64 (2x4 16x16 accs).
// Double-buffered LDS; tiles streamed with GLOBAL_LOAD_ASYNC_TO_LDS_B128
// (ASYNCcnt) so tile kt+1 loads overlap WMMA compute of tile kt.
// ---------------------------------------------------------------------------
__global__ __launch_bounds__(256) void gemm128_kernel(
    const _Float16* __restrict__ ehi, const _Float16* __restrict__ elo,
    float* __restrict__ S) {
  __shared__ _Float16 Ah[2][TILE * LDK];
  __shared__ _Float16 Al[2][TILE * LDK];
  __shared__ _Float16 Bh[2][TILE * LDK];
  __shared__ _Float16 Bl[2][TILE * LDK];

  const int t = threadIdx.x;
  const int i0 = blockIdx.x * TILE;
  const int j0 = blockIdx.y * TILE;
  const int lane = t & 31;
  const int wid = t >> 5;
  const int wr = wid & 3;   // 4 wave-rows of 32
  const int wc = wid >> 2;  // 2 wave-cols of 64

  const v8f zf = {0.f, 0.f, 0.f, 0.f, 0.f, 0.f, 0.f, 0.f};
  v8f acc[2][4];
#pragma unroll
  for (int rt = 0; rt < 2; ++rt)
#pragma unroll
    for (int ct = 0; ct < 4; ++ct) acc[rt][ct] = zf;

  const int ldrow = t >> 2;       // 0..63
  const int ldch = (t & 3) * 8;   // half-offset of 16B chunk
  const int m = lane & 15;
  const int kbA = (lane < 16) ? 0 : 8;
  const int kbB = (lane < 16) ? 0 : 8;

  // stage one 128x32 sub-tile set (Ah/Al/Bh/Bl) for K-block kt into buffer b
  auto issue_tile = [&](int kt, int b) {
    const int K0 = kt * 32;
#pragma unroll
    for (int it = 0; it < 2; ++it) {
      const int row = ldrow + it * 64;
      const size_t ga = (size_t)(i0 + row) * ISIZE + K0 + ldch;
      const size_t gb = (size_t)(j0 + row) * ISIZE + K0 + ldch;
      const int lo = row * LDK + ldch;
#ifdef HAVE_ASYNC_LDS
      __builtin_amdgcn_global_load_async_to_lds_b128(
          (v4i*)&ehi[ga], (v4i*)&Ah[b][lo], 0, 0);
      __builtin_amdgcn_global_load_async_to_lds_b128(
          (v4i*)&elo[ga], (v4i*)&Al[b][lo], 0, 0);
      __builtin_amdgcn_global_load_async_to_lds_b128(
          (v4i*)&ehi[gb], (v4i*)&Bh[b][lo], 0, 0);
      __builtin_amdgcn_global_load_async_to_lds_b128(
          (v4i*)&elo[gb], (v4i*)&Bl[b][lo], 0, 0);
#else
      *(uint4*)&Ah[b][lo] = *(const uint4*)&ehi[ga];
      *(uint4*)&Al[b][lo] = *(const uint4*)&elo[ga];
      *(uint4*)&Bh[b][lo] = *(const uint4*)&ehi[gb];
      *(uint4*)&Bl[b][lo] = *(const uint4*)&elo[gb];
#endif
    }
  };

  issue_tile(0, 0);  // prologue: tile 0 in flight

#pragma unroll 1
  for (int kt = 0; kt < KTILES; ++kt) {
    const int cur = kt & 1;
    if (kt + 1 < KTILES) {
      // buffer cur^1 held tile kt-1; all waves finished it before the barrier
      // that ended iteration kt-1, so overwriting is safe.
      issue_tile(kt + 1, cur ^ 1);
#ifdef HAVE_ASYNC_LDS
      wait_async_le8();  // in-order ASYNCcnt: tile kt's 8 loads have landed
#endif
    } else {
#ifdef HAVE_ASYNC_LDS
      wait_async_le0();
#endif
    }
    __syncthreads();  // all waves' tile-kt data visible

    v16h ah[2], al[2];
#pragma unroll
    for (int rt = 0; rt < 2; ++rt) {
      const int r = wr * 32 + rt * 16 + m;
      ah[rt] = frag_a(&Ah[cur][r * LDK + kbA]);
      al[rt] = frag_a(&Al[cur][r * LDK + kbA]);
    }
#pragma unroll
    for (int ct = 0; ct < 4; ++ct) {
      const int n = wc * 64 + ct * 16 + m;
      const v16h bh = frag_b(&Bh[cur][n * LDK + kbB]);
      const v16h bl = frag_b(&Bl[cur][n * LDK + kbB]);
#pragma unroll
      for (int rt = 0; rt < 2; ++rt) {
        acc[rt][ct] = __builtin_amdgcn_wmma_f32_16x16x32_f16(
            false, ah[rt], false, bh, (short)0, acc[rt][ct], false, false);
        acc[rt][ct] = __builtin_amdgcn_wmma_f32_16x16x32_f16(
            false, ah[rt], false, bl, (short)0, acc[rt][ct], false, false);
        acc[rt][ct] = __builtin_amdgcn_wmma_f32_16x16x32_f16(
            false, al[rt], false, bh, (short)0, acc[rt][ct], false, false);
      }
    }
    __syncthreads();  // everyone done with buffer cur before it is re-filled
  }

  // C/D layout: VGPR r -> M = r (+8 for lanes 16..31); N = lane & 15
  const int mofs = (lane >> 4) * 8;
  const int nloc = lane & 15;
#pragma unroll
  for (int rt = 0; rt < 2; ++rt)
#pragma unroll
    for (int ct = 0; ct < 4; ++ct) {
      const int col = j0 + wc * 64 + ct * 16 + nloc;
#pragma unroll
      for (int r = 0; r < 8; ++r) {
        const int row = i0 + wr * 32 + rt * 16 + mofs + r;
        if (row < NROWS && col < NROWS)
          S[(size_t)row * NROWS + col] = acc[rt][ct][r];
      }
    }
}

// ---------------------------------------------------------------------------
// Kernel 3: per-row top-31 mask + relu, in place. Block per row; row in LDS.
// ---------------------------------------------------------------------------
__global__ __launch_bounds__(256) void topk_relu_kernel(float* __restrict__ S) {
  __shared__ float row[NROWS];
  __shared__ float rv[256];
  __shared__ int ri[256];
  __shared__ float topv[KSEL];
  __shared__ int topi[KSEL];
  const int t = threadIdx.x;
  const size_t base = (size_t)blockIdx.x * NROWS;

  for (int j = t; j < NROWS; j += 256) row[j] = S[base + j];
  __syncthreads();

#pragma unroll 1
  for (int it = 0; it < KSEL; ++it) {
    float v = -__builtin_inff();
    int idx = 0x7fffffff;
    for (int j = t; j < NROWS; j += 256)
      if (row[j] > v) { v = row[j]; idx = j; }  // strict > keeps smallest j
    rv[t] = v;
    ri[t] = idx;
    __syncthreads();
#pragma unroll 1
    for (int s = 128; s > 0; s >>= 1) {
      if (t < s) {
        const float ov = rv[t + s];
        const int oi = ri[t + s];
        if (ov > rv[t] || (ov == rv[t] && oi < ri[t])) { rv[t] = ov; ri[t] = oi; }
      }
      __syncthreads();
    }
    if (t == 0) {
      topv[it] = rv[0];
      topi[it] = ri[0];
      row[ri[0]] = -__builtin_inff();
    }
    __syncthreads();
  }

  const float thresh = topv[KSEL - 1];
  if (t < KSEL) row[topi[t]] = topv[t];  // restore extracted entries
  __syncthreads();

  for (int j = t; j < NROWS; j += 256) {
    const float v = row[j];
    S[base + j] = (v >= thresh && v > 0.f) ? v : 0.f;
  }
}

// ---------------------------------------------------------------------------
extern "C" void kernel_launch(void* const* d_in, const int* in_sizes, int n_in,
                              void* d_out, int out_size, void* d_ws,
                              size_t ws_size, hipStream_t stream) {
  const float* feat = (const float*)d_in[0];
  const float* W1 = (const float*)d_in[1];
  const float* b1 = (const float*)d_in[2];
  const float* W2 = (const float*)d_in[3];
  const float* b2 = (const float*)d_in[4];
  float* S = (float*)d_out;

  char* ws = (char*)d_ws;
  _Float16* ehi = (_Float16*)ws;
  _Float16* elo = (_Float16*)(ws + (size_t)NPAD * ISIZE * sizeof(_Float16));

  mlp_norm_kernel<<<NROWS / 16 + 1, 256, 0, stream>>>(feat, W1, b1, W2, b2,
                                                      ehi, elo);
  dim3 grid(NTILE, NTILE);
  gemm128_kernel<<<grid, 256, 0, stream>>>(ehi, elo, S);
  topk_relu_kernel<<<NROWS, 256, 0, stream>>>(S);
}
